// AttnBlock_28544352649733
// MI455X (gfx1250) — matrix operand order repaired and checked by reading, not compiled, not run on hardware
//
#include <hip/hip_runtime.h>

// ---------------------------------------------------------------------------
// AttnBlock for MI455X (gfx1250): bf16 WMMA pipeline, fp32 accumulate.
// B=4, C=256, H=W=64, N=4096, 32 groups.
// ---------------------------------------------------------------------------

typedef __attribute__((ext_vector_type(16))) __bf16 v16bf;
typedef __attribute__((ext_vector_type(8)))  float  v8f;

#define B_  4
#define C_  256
#define N_  4096
#define CN_ (C_ * N_)          // 1048576 elements per batch
#define S_STRIDE 4100          // padded LDS score-row stride (banks: row*4+e)

static __device__ __forceinline__ v8f wmma_bf16(v16bf a, v16bf b, v8f c) {
  // D = A(16x32) * B(32x16) + C(16x16), fp32 accum
  return __builtin_amdgcn_wmma_f32_16x16x32_bf16(false, a, false, b, (short)0, c,
                                                 false, false);
}

static __device__ __forceinline__ v16bf load_frag_2x16B(const __bf16* p0,
                                                        const __bf16* p1) {
  union { uint4 u[2]; v16bf v; } r;
  r.u[0] = *(const uint4*)p0;   // elements 0..7
  r.u[1] = *(const uint4*)p1;   // elements 8..15
  return r.v;
}

// A fragment: 16(M) x 32(K) bf16, memory row-major [M][K], leading dim lda.
// Lane (r = lane&15, hi = lane>>4) holds K = kb + hi*16 .. +15 (contiguous).
static __device__ __forceinline__ v16bf load_a_frag(const __bf16* A, int lda,
                                                    int kb, int lane) {
  int r = lane & 15, hi = lane >> 4;
  const __bf16* p = A + r * lda + kb + hi * 16;
  return load_frag_2x16B(p, p + 8);
}

// B fragment: 32(K) x 16(N) bf16, memory stored N-major [N][K], leading dim ldb.
// Lane (n = lane&15, hi = lane>>4) holds K = kb + hi*16 .. +15 (contiguous).
static __device__ __forceinline__ v16bf load_b_frag(const __bf16* Bt, int ldb,
                                                    int kb, int lane) {
  int n = lane & 15, hi = lane >> 4;
  const __bf16* p = Bt + n * ldb + kb + hi * 16;
  return load_frag_2x16B(p, p + 8);
}

// ---------------------------------------------------------------------------
// K0: convert the four 256x256 fp32 weight matrices to bf16 (row-major kept).
// ---------------------------------------------------------------------------
__global__ void ab_wcvt_kernel(const float* Wq, const float* Wk, const float* Wv,
                               const float* Wo, __bf16* wq, __bf16* wk,
                               __bf16* wv, __bf16* wo) {
  int i = blockIdx.x * 256 + threadIdx.x;   // 0..65535
  wq[i] = (__bf16)Wq[i];
  wk[i] = (__bf16)Wk[i];
  wv[i] = (__bf16)Wv[i];
  wo[i] = (__bf16)Wo[i];
}

// ---------------------------------------------------------------------------
// K1: GroupNorm statistics. One WG per (b, group); group = 8 channels * 4096
// spatial = 32768 contiguous floats. Output (mean, rstd).
// ---------------------------------------------------------------------------
__global__ void ab_gn_stats_kernel(const float* __restrict__ x,
                                   float2* __restrict__ stats) {
  int bg = blockIdx.x;                               // 0..127
  const float* base = x + (size_t)bg * 32768;
  float s = 0.f, ss = 0.f;
  for (int e = threadIdx.x; e < 32768; e += 256) {
    float v = base[e];
    s += v;
    ss += v * v;
  }
  __shared__ float rs[256], rss[256];
  rs[threadIdx.x] = s;
  rss[threadIdx.x] = ss;
  __syncthreads();
  for (int st = 128; st > 0; st >>= 1) {
    if (threadIdx.x < st) {
      rs[threadIdx.x] += rs[threadIdx.x + st];
      rss[threadIdx.x] += rss[threadIdx.x + st];
    }
    __syncthreads();
  }
  if (threadIdx.x == 0) {
    float m = rs[0] * (1.0f / 32768.0f);
    float var = rss[0] * (1.0f / 32768.0f) - m * m;
    stats[bg] = make_float2(m, rsqrtf(var + 1e-6f));
  }
}

// ---------------------------------------------------------------------------
// K2: apply GroupNorm and write hn transposed to bf16 [b][n][c] (N-major,
// which is the B-fragment-friendly layout for the QKV GEMM). Each 8-channel
// group is packed into one 16B store.
// ---------------------------------------------------------------------------
__global__ void ab_gn_apply_kernel(const float* __restrict__ x,
                                   const float* __restrict__ gamma,
                                   const float* __restrict__ beta,
                                   const float2* __restrict__ stats,
                                   __bf16* __restrict__ hn_t) {
  int gid = blockIdx.x * 256 + threadIdx.x;   // 0..B_*N_-1
  int b = gid >> 12;
  int n = gid & (N_ - 1);
  const float* xb = x + (size_t)b * CN_;
  __bf16* dst = hn_t + (size_t)gid * C_;
  for (int g = 0; g < 32; g++) {
    float2 st = stats[b * 32 + g];
    union { __bf16 h[8]; uint4 u; } pk;
#pragma unroll
    for (int cc = 0; cc < 8; cc++) {
      int c = g * 8 + cc;
      float v = (xb[(size_t)c * N_ + n] - st.x) * st.y * gamma[c] + beta[c];
      pk.h[cc] = (__bf16)v;
    }
    *(uint4*)(dst + g * 8) = pk.u;
  }
}

// ---------------------------------------------------------------------------
// K3: fused QKV projection GEMM. One wave per 16x16 output tile, K=256.
// The hn B-fragment is loaded once per K-chunk and reused for Wq/Wk/Wv.
// Outputs: q_t [b][n][c] (scale c^-0.5 folded in), k_t [b][n][c], v [b][c][n].
// ---------------------------------------------------------------------------
__global__ __launch_bounds__(256)
void ab_qkv_kernel(const __bf16* __restrict__ hn_t, const __bf16* __restrict__ wq,
                   const __bf16* __restrict__ wk, const __bf16* __restrict__ wv,
                   const float* __restrict__ bq, const float* __restrict__ bk,
                   const float* __restrict__ bv, __bf16* __restrict__ q_t,
                   __bf16* __restrict__ k_t, __bf16* __restrict__ vmat) {
  int wave = (blockIdx.x * 256 + threadIdx.x) >> 5;   // 0..16383
  int lane = threadIdx.x & 31;
  int nt = wave & 255;          // n-tile (0..255)
  int ct = (wave >> 8) & 15;    // c_out tile (0..15)
  int b = wave >> 12;           // batch

  const __bf16* hb = hn_t + (size_t)b * CN_ + (size_t)nt * 16 * C_;
  const __bf16* aq = wq + ct * 16 * C_;
  const __bf16* ak = wk + ct * 16 * C_;
  const __bf16* av = wv + ct * 16 * C_;

  v8f accq = {0, 0, 0, 0, 0, 0, 0, 0};
  v8f acck = {0, 0, 0, 0, 0, 0, 0, 0};
  v8f accv = {0, 0, 0, 0, 0, 0, 0, 0};
#pragma unroll
  for (int kc = 0; kc < 8; kc++) {
    int kb = kc * 32;
    v16bf fb = load_b_frag(hb, C_, kb, lane);
    accq = wmma_bf16(load_a_frag(aq, C_, kb, lane), fb, accq);
    acck = wmma_bf16(load_a_frag(ak, C_, kb, lane), fb, acck);
    accv = wmma_bf16(load_a_frag(av, C_, kb, lane), fb, accv);
  }

  int col_n = nt * 16 + (lane & 15);
  int mbase = (lane >> 4) * 8;
  int co0 = ct * 16 + mbase;
  const float scale = 0.0625f;   // 256^-0.5 folded into q

  union { __bf16 h[8]; uint4 u; } pq, pk;
  __bf16* vrow = vmat + (size_t)b * CN_ + col_n;
#pragma unroll
  for (int m = 0; m < 8; m++) {
    int co = co0 + m;
    pq.h[m] = (__bf16)((accq[m] + bq[co]) * scale);
    pk.h[m] = (__bf16)(acck[m] + bk[co]);
    vrow[(size_t)co * N_] = (__bf16)(accv[m] + bv[co]);
  }
  *(uint4*)(q_t + (size_t)b * CN_ + (size_t)col_n * C_ + co0) = pq.u;
  *(uint4*)(k_t + (size_t)b * CN_ + (size_t)col_n * C_ + co0) = pk.u;
}

// ---------------------------------------------------------------------------
// K4: attention. One WG per (batch, 16-row i-tile). The full 16x4096 fp32
// score slab lives in LDS (263 KB of the WGP's 320 KB, padded rows to kill
// bank conflicts). 8 waves:
//   Phase A: QK^T via WMMA (q rows preloaded, reused across 32 j-tiles/wave)
//   Phase B: exact row softmax in LDS (denominator folded into PV epilogue)
//   Phase C: PV via WMMA; A = v (natural [c][n]), B = attn^T read as
//            contiguous fp32 from LDS and converted to bf16 in registers
//            (the cvts co-execute with the multicycle XDL WMMA).
// Output written transposed: out_t [b][n][c].
// ---------------------------------------------------------------------------
__global__ __launch_bounds__(256)
void ab_attn_kernel(const __bf16* __restrict__ q_t, const __bf16* __restrict__ k_t,
                    const __bf16* __restrict__ vmat, __bf16* __restrict__ out_t) {
  extern __shared__ float lds[];
  float* S = lds;                      // 16 rows * S_STRIDE
  float* red = lds + 16 * S_STRIDE;    // 16 * 16 partial reductions
  float* rowstat = red + 256;          // 16: rowmax, then 1/rowsum

  int b = blockIdx.x >> 8;
  int it = blockIdx.x & 255;
  int tid = threadIdx.x;
  int wid = tid >> 5;
  int lane = tid & 31;

  const __bf16* qb = q_t + (size_t)b * CN_ + (size_t)it * 16 * C_;
  const __bf16* kb = k_t + (size_t)b * CN_;

  // Preload this i-tile's q A-fragments for all 8 K-chunks (reused 32x).
  v16bf afrag[8];
#pragma unroll
  for (int kc = 0; kc < 8; kc++) afrag[kc] = load_a_frag(qb, C_, kc * 32, lane);

  // ---- Phase A: scores = q^T k for 32 j-tiles per wave ----
  for (int jt = wid; jt < 256; jt += 8) {
    const __bf16* kt = kb + (size_t)jt * 16 * C_;
    if (jt + 8 < 256) {   // hint next panel into cache (global_prefetch_b8)
      __builtin_prefetch(kt + 8 * 16 * C_ + (lane & 15) * C_, 0, 0);
      __builtin_prefetch(kt + 8 * 16 * C_ + (lane & 15) * C_ + 128, 0, 0);
    }
    v8f acc = {0, 0, 0, 0, 0, 0, 0, 0};
#pragma unroll
    for (int kc = 0; kc < 8; kc++)
      acc = wmma_bf16(afrag[kc], load_b_frag(kt, C_, kc * 32, lane), acc);
    int j = jt * 16 + (lane & 15);
    int mbase = (lane >> 4) * 8;
#pragma unroll
    for (int m = 0; m < 8; m++) S[(mbase + m) * S_STRIDE + j] = acc[m];
  }
  __syncthreads();

  // ---- Phase B: softmax (max, exp, sum); keep exp values in LDS fp32 ----
  {
    int r = tid >> 4, seg = tid & 15;
    float* row = S + r * S_STRIDE + seg * 256;
    float mx = -1e30f;
    for (int e = 0; e < 256; e++) mx = fmaxf(mx, row[e]);
    red[r * 16 + seg] = mx;
    __syncthreads();
    if (seg == 0) {
      float m = red[r * 16];
      for (int e = 1; e < 16; e++) m = fmaxf(m, red[r * 16 + e]);
      rowstat[r] = m;
    }
    __syncthreads();
    float m = rowstat[r];
    float sm = 0.f;
    for (int e = 0; e < 256; e++) {
      float ev = __expf(row[e] - m);
      row[e] = ev;
      sm += ev;
    }
    __syncthreads();              // all reads of rowstat (max) done
    red[r * 16 + seg] = sm;
    __syncthreads();
    if (seg == 0) {
      float s2 = 0.f;
      for (int e = 0; e < 16; e++) s2 += red[r * 16 + e];
      rowstat[r] = 1.0f / s2;     // inverse denominator for PV epilogue
    }
    __syncthreads();
  }

  // ---- Phase C: out[c,i] = sum_j v[c,j] * attn[i,j]; 2 c-tiles per wave ----
  const __bf16* vb = vmat + (size_t)b * CN_;
  int icol = lane & 15, hi = lane >> 4;
  for (int ct = wid; ct < 16; ct += 8) {
    const __bf16* va = vb + (size_t)ct * 16 * N_;
    v8f acc = {0, 0, 0, 0, 0, 0, 0, 0};
#pragma unroll 2
    for (int kc = 0; kc < 128; kc++) {
      int kb2 = kc * 32;
      if (kc + 2 < 128)   // hint next v chunk (global_prefetch_b8)
        __builtin_prefetch(va + (lane & 15) * N_ + kb2 + 64, 0, 0);
      v16bf af = load_a_frag(va, N_, kb2, lane);           // v rows, contiguous
      const float* p = S + icol * S_STRIDE + kb2 + hi * 16; // attn^T, contiguous
      v16bf bf;
#pragma unroll
      for (int e = 0; e < 16; e++) bf[e] = (__bf16)p[e];
      acc = wmma_bf16(af, bf, acc);
    }
    float inv = rowstat[icol];
    int mbase = hi * 8;
    union { __bf16 h[8]; uint4 u; } po;
#pragma unroll
    for (int m = 0; m < 8; m++) po.h[m] = (__bf16)(acc[m] * inv);
    int n = it * 16 + icol;
    *(uint4*)(out_t + (size_t)b * CN_ + (size_t)n * C_ + ct * 16 + mbase) = po.u;
  }
}

// ---------------------------------------------------------------------------
// K5: output projection y = Wo * out + bo + x (residual). One wave per tile.
// ---------------------------------------------------------------------------
__global__ __launch_bounds__(256)
void ab_proj_kernel(const __bf16* __restrict__ wo, const float* __restrict__ bo,
                    const __bf16* __restrict__ out_t, const float* __restrict__ x,
                    float* __restrict__ y) {
  int wave = (blockIdx.x * 256 + threadIdx.x) >> 5;
  int lane = threadIdx.x & 31;
  int nt = wave & 255;
  int ct = (wave >> 8) & 15;
  int b = wave >> 12;

  const __bf16* ob = out_t + (size_t)b * CN_ + (size_t)nt * 16 * C_;
  const __bf16* ao = wo + ct * 16 * C_;
  v8f acc = {0, 0, 0, 0, 0, 0, 0, 0};
#pragma unroll
  for (int kc = 0; kc < 8; kc++) {
    int kb = kc * 32;
    acc = wmma_bf16(load_a_frag(ao, C_, kb, lane), load_b_frag(ob, C_, kb, lane),
                    acc);
  }
  int n = nt * 16 + (lane & 15);
  int c0 = ct * 16 + (lane >> 4) * 8;
#pragma unroll
  for (int m = 0; m < 8; m++) {
    int c = c0 + m;
    size_t idx = (size_t)b * CN_ + (size_t)c * N_ + n;
    y[idx] = acc[m] + bo[c] + x[idx];
  }
}

// ---------------------------------------------------------------------------
// Launch
// ---------------------------------------------------------------------------
extern "C" void kernel_launch(void* const* d_in, const int* in_sizes, int n_in,
                              void* d_out, int out_size, void* d_ws, size_t ws_size,
                              hipStream_t stream) {
  const float* x     = (const float*)d_in[0];
  const float* gamma = (const float*)d_in[1];
  const float* beta  = (const float*)d_in[2];
  const float* Wq    = (const float*)d_in[3];
  const float* bq    = (const float*)d_in[4];
  const float* Wk    = (const float*)d_in[5];
  const float* bk    = (const float*)d_in[6];
  const float* Wv    = (const float*)d_in[7];
  const float* bv    = (const float*)d_in[8];
  const float* Wo    = (const float*)d_in[9];
  const float* bo    = (const float*)d_in[10];
  float* out = (float*)d_out;

  char* ws = (char*)d_ws;
  size_t off = 0;
  float2* stats = (float2*)(ws + off); off += 4096;
  const size_t act_bytes = (size_t)B_ * CN_ * 2;   // bf16 activation buffer
  __bf16* hn_t  = (__bf16*)(ws + off); off += act_bytes;
  __bf16* q_t   = (__bf16*)(ws + off); off += act_bytes;
  __bf16* k_t   = (__bf16*)(ws + off); off += act_bytes;
  __bf16* vmat  = (__bf16*)(ws + off); off += act_bytes;
  __bf16* out_t = (__bf16*)(ws + off); off += act_bytes;
  __bf16* wq = (__bf16*)(ws + off); off += C_ * C_ * 2;
  __bf16* wk = (__bf16*)(ws + off); off += C_ * C_ * 2;
  __bf16* wv = (__bf16*)(ws + off); off += C_ * C_ * 2;
  __bf16* wo = (__bf16*)(ws + off); off += C_ * C_ * 2;
  (void)ws_size; (void)in_sizes; (void)n_in; (void)out_size;

  size_t attn_lds = (size_t)(16 * S_STRIDE + 256 + 16) * sizeof(float); // ~258 KB
  hipFuncSetAttribute((const void*)ab_attn_kernel,
                      hipFuncAttributeMaxDynamicSharedMemorySize, (int)attn_lds);

  ab_wcvt_kernel<<<256, 256, 0, stream>>>(Wq, Wk, Wv, Wo, wq, wk, wv, wo);
  ab_gn_stats_kernel<<<128, 256, 0, stream>>>(x, stats);
  ab_gn_apply_kernel<<<64, 256, 0, stream>>>(x, gamma, beta, stats, hn_t);
  ab_qkv_kernel<<<2048, 256, 0, stream>>>(hn_t, wq, wk, wv, bq, bk, bv, q_t, k_t,
                                          vmat);
  ab_attn_kernel<<<B_ * 256, 256, attn_lds, stream>>>(q_t, k_t, vmat, out_t);
  ab_proj_kernel<<<2048, 256, 0, stream>>>(wo, bo, out_t, x, out);
}
